// HorizonMultiheadAttention_83021717832136
// MI455X (gfx1250) — matrix operand order repaired and checked by reading, not compiled, not run on hardware
//
#include <hip/hip_runtime.h>

typedef __attribute__((ext_vector_type(16))) __bf16 v16bf;
typedef __attribute__((ext_vector_type(8)))  __bf16 v8bf;
typedef __attribute__((ext_vector_type(4)))  __bf16 v4bf;
typedef __attribute__((ext_vector_type(8)))  float  v8f;
typedef __attribute__((ext_vector_type(4)))  unsigned int u32x4;
typedef __attribute__((ext_vector_type(8)))  int         i32x8;
typedef __attribute__((ext_vector_type(4)))  int         i32x4;

#define EMB    256
#define SEQ    1024
#define NHEADS 8
#define HDIM   32

// Assemble a v16bf fragment from two contiguous 8x-bf16 (16B) runs.
__device__ __forceinline__ v16bf ldfrag(const __bf16* p0, const __bf16* p1) {
  v8bf lo = *(const v8bf*)p0;
  v8bf hi = *(const v8bf*)p1;
  v16bf r;
#pragma unroll
  for (int i = 0; i < 8; ++i) { r[i] = lo[i]; r[i + 8] = hi[i]; }
  return r;
}

// ---------------------------------------------------------------------------
// Tiled GEMM:  Y[b][o][t] = sum_i W[o][i] * X[b][i][t] + bias[o]
// Block = 256 threads (8 waves). Block tile M=16 x N=512; each wave owns
// N=64 (4 accumulators, A reused 4x -> 4 WMMAs per K step). K step 32.
// mode 0: bf16 out, layout [bh][t][32]  (Q, pre-scaled by alpha)
// mode 1: bf16 out, layout [bh][s][32]  (K transposed)
// mode 2: bf16 out, layout [b][e][s]    (V natural)
// mode 3: f32  out, layout [b][e][t]    (final output)
// ---------------------------------------------------------------------------
__global__ __launch_bounds__(256) void proj_gemm(
    const float* __restrict__ W, const float* __restrict__ bias,
    const void* __restrict__ Xin, int in_bf16, float alpha, int mode,
    void* __restrict__ Out)
{
  __shared__ __bf16 lds_a[16 * 32];    // [m][k]
  __shared__ __bf16 lds_b[512 * 32];   // [t][k]  (B-frag cols contiguous in k)

  const int tid  = threadIdx.x;
  const int lane = tid & 31, wave = tid >> 5;
  const int half = lane >> 4, nlo = lane & 15;
  const int b  = blockIdx.z;
  const int o0 = blockIdx.y * 16;
  const int t0 = blockIdx.x * 512;

  v8f acc[4] = {};
  for (int k0 = 0; k0 < EMB; k0 += 32) {
    { // stage W tile [16 o][32 k] -> bf16 (native converts)
      int idx = tid * 2;
      int m = idx >> 5, kk = idx & 31;
      float2 w2 = *(const float2*)(W + (size_t)(o0 + m) * EMB + k0 + kk);
      lds_a[m * 32 + kk]     = (__bf16)w2.x;
      lds_a[m * 32 + kk + 1] = (__bf16)w2.y;
    }
    // stage X tile [32 k][512 t] transposed into lds_b[t][k] via 4x4 blocks:
    // 4 float4 row loads -> 4 packed v4bf (8B) column stores.
    if (!in_bf16) {
      const float* X = (const float*)Xin;
#pragma unroll
      for (int i = 0; i < 4; ++i) {
        int blk = tid * 4 + i;               // 1024 4x4 blocks
        int tb = blk & 127, kb = blk >> 7;
        const float* src = X + ((size_t)b * EMB + k0 + kb * 4) * SEQ + t0 + tb * 4;
        float4 r0 = *(const float4*)(src);
        float4 r1 = *(const float4*)(src + SEQ);
        float4 r2 = *(const float4*)(src + 2 * SEQ);
        float4 r3 = *(const float4*)(src + 3 * SEQ);
        __bf16* dst = lds_b + (tb * 4) * 32 + kb * 4;
        v4bf c0 = {(__bf16)r0.x, (__bf16)r1.x, (__bf16)r2.x, (__bf16)r3.x};
        v4bf c1 = {(__bf16)r0.y, (__bf16)r1.y, (__bf16)r2.y, (__bf16)r3.y};
        v4bf c2 = {(__bf16)r0.z, (__bf16)r1.z, (__bf16)r2.z, (__bf16)r3.z};
        v4bf c3 = {(__bf16)r0.w, (__bf16)r1.w, (__bf16)r2.w, (__bf16)r3.w};
        *(v4bf*)(dst)      = c0;
        *(v4bf*)(dst + 32) = c1;
        *(v4bf*)(dst + 64) = c2;
        *(v4bf*)(dst + 96) = c3;
      }
    } else {
      const __bf16* X = (const __bf16*)Xin;
#pragma unroll
      for (int i = 0; i < 4; ++i) {
        int blk = tid * 4 + i;
        int tb = blk & 127, kb = blk >> 7;
        const __bf16* src = X + ((size_t)b * EMB + k0 + kb * 4) * SEQ + t0 + tb * 4;
        v4bf r0 = *(const v4bf*)(src);
        v4bf r1 = *(const v4bf*)(src + SEQ);
        v4bf r2 = *(const v4bf*)(src + 2 * SEQ);
        v4bf r3 = *(const v4bf*)(src + 3 * SEQ);
        __bf16* dst = lds_b + (tb * 4) * 32 + kb * 4;
        v4bf c0 = {r0[0], r1[0], r2[0], r3[0]};
        v4bf c1 = {r0[1], r1[1], r2[1], r3[1]};
        v4bf c2 = {r0[2], r1[2], r2[2], r3[2]};
        v4bf c3 = {r0[3], r1[3], r2[3], r3[3]};
        *(v4bf*)(dst)      = c0;
        *(v4bf*)(dst + 32) = c1;
        *(v4bf*)(dst + 64) = c2;
        *(v4bf*)(dst + 96) = c3;
      }
    }
    __syncthreads();

    const int ka = half * 8;               // unused; A-frag K base is half * 16 (lanes>=16: +16)
    v16bf a = ldfrag(lds_a + nlo * 32 + half * 16,
                     lds_a + nlo * 32 + half * 16 + 8);
#pragma unroll
    for (int n = 0; n < 4; ++n) {
      const __bf16* brow = lds_b + ((size_t)(wave * 64 + n * 16 + nlo)) * 32 + half * 16;
      v16bf bb = ldfrag(brow, brow + 8);
      acc[n] = __builtin_amdgcn_wmma_f32_16x16x32_bf16(false, a, false, bb,
                                                       (short)0, acc[n], false, false);
    }
    __syncthreads();
  }

#pragma unroll
  for (int n = 0; n < 4; ++n) {
    const int tcol = t0 + wave * 64 + n * 16 + nlo;
    float vals[8];
#pragma unroll
    for (int j = 0; j < 8; ++j)
      vals[j] = (acc[n][j] + bias[o0 + j + 8 * half]) * alpha;

    if (mode == 0 || mode == 1) {          // [bh][t][32] bf16, packed 16B store
      __bf16* outb = (__bf16*)Out;
      const int head = o0 >> 5;
      const int bh   = b * NHEADS + head;
      const int h0   = (o0 & 31) + half * 8;
      v8bf pk;
#pragma unroll
      for (int j = 0; j < 8; ++j) pk[j] = (__bf16)vals[j];
      *(v8bf*)(outb + ((size_t)bh * SEQ + tcol) * HDIM + h0) = pk;
    } else if (mode == 2) {                // [b][e][s] bf16
      __bf16* outb = (__bf16*)Out;
#pragma unroll
      for (int j = 0; j < 8; ++j)
        outb[((size_t)b * EMB + o0 + j + 8 * half) * SEQ + tcol] = (__bf16)vals[j];
    } else {                               // [b][e][t] f32 final
      float* outf = (float*)Out;
#pragma unroll
      for (int j = 0; j < 8; ++j)
        outf[((size_t)b * EMB + o0 + j + 8 * half) * SEQ + tcol] = vals[j];
    }
  }
}

// ---------------------------------------------------------------------------
// Attention: 8 waves/block share a TDM-loaded K panel (64KB in LDS); each
// wave owns one 16-row t-tile. Pass 1: rowmax(QK^T + mask). Pass 2: exp,
// rowsum, unnormalized O via PV WMMAs (P routed through a per-wave LDS tile),
// then 1/rowsum scale. Dynamic LDS: 32768 bf16 K panel + 8 * 512 bf16 P.
// ---------------------------------------------------------------------------
__global__ __launch_bounds__(256) void attn_kernel(
    const __bf16* __restrict__ qbuf,  // [bh][t][32], pre-scaled
    const __bf16* __restrict__ kbuf,  // [bh][s][32]
    const __bf16* __restrict__ vbuf,  // [b][e][s]
    const float*  __restrict__ mask,  // [T][S]
    __bf16*       __restrict__ abuf)  // [b][e][t]
{
  extern __shared__ __bf16 smem[];
  __bf16* k_panel = smem;                       // [1024][32]
  const int tid  = threadIdx.x;
  const int lane = tid & 31, wave = tid >> 5;
  __bf16* p_tile = smem + 32768 + wave * 512;   // per-wave [16][32]

  const int half = lane >> 4, nlo = lane & 15;
  const int bh = blockIdx.y;
  const int b = bh >> 3, head = bh & 7;
  const int t0 = blockIdx.x * 128 + wave * 16;
  const int ka = half * 8;

  // ---- TDM: DMA K panel (1024x32 bf16 = 64KB) into LDS (wave 0 issues) ----
  if (wave == 0) {
    unsigned lds_off = (unsigned)(size_t)(void*)k_panel;   // low 32b = LDS byte offset
    unsigned long long ga =
        (unsigned long long)(size_t)(const void*)(kbuf + (size_t)bh * SEQ * HDIM);
    u32x4 g0; i32x8 g1; i32x4 g2; i32x4 g3; i32x8 g4;
    g0[0] = 1u;                                  // count=1 (valid user D#)
    g0[1] = lds_off;                             // lds_addr
    g0[2] = (unsigned)ga;                        // global_addr[31:0]
    g0[3] = (unsigned)(ga >> 32) | (2u << 30);   // global_addr[56:32] | type=2
    g1[0] = (1 << 16);                           // data_size=2B, wg_mask=0
    g1[1] = (int)(0x8000u << 16);                // tensor_dim0 = 32768 (bits 63:48)
    g1[2] = (1 << 16);                           // tensor_dim1 = 1    (bits 95:80)
    g1[3] = (int)(0x8000u << 16);                // tile_dim0  = 32768 (bits 127:112)
    g1[4] = 1;                                   // tile_dim1 = 1
    g1[5] = 32768;                               // tensor_dim0_stride
    g1[6] = 0;
    g1[7] = 0;
    g2[0] = g2[1] = g2[2] = g2[3] = 0;
    g3[0] = g3[1] = g3[2] = g3[3] = 0;
    g4[0] = g4[1] = g4[2] = g4[3] = 0;
    g4[4] = g4[5] = g4[6] = g4[7] = 0;
    __builtin_amdgcn_tensor_load_to_lds(g0, g1, g2, g3, g4, 0);
    __builtin_amdgcn_s_wait_tensorcnt((short)0);
  }
  __syncthreads();

  // Q A-fragment: row M = nlo, contiguous hd run (global, 2x16B loads)
  const __bf16* qrow = qbuf + ((size_t)bh * SEQ + t0 + nlo) * HDIM;
  v16bf aq = ldfrag(qrow + half * 16, qrow + half * 16 + 8);

  // ---- pass 1: rowmax ----
  float m8[8];
#pragma unroll
  for (int j = 0; j < 8; ++j) m8[j] = -1e30f;

  for (int s0 = 0; s0 < SEQ; s0 += 16) {
    const __bf16* krow = k_panel + (size_t)(s0 + nlo) * HDIM + half * 16;
    v16bf bk = ldfrag(krow, krow + 8);
    v8f c = {};
    c = __builtin_amdgcn_wmma_f32_16x16x32_bf16(false, aq, false, bk,
                                                (short)0, c, false, false);
    const float* mp = mask + (size_t)(t0 + half * 8) * SEQ + s0 + nlo;
#pragma unroll
    for (int j = 0; j < 8; ++j)
      m8[j] = fmaxf(m8[j], c[j] + mp[(size_t)j * SEQ]);
  }
#pragma unroll
  for (int d = 1; d < 16; d <<= 1)
#pragma unroll
    for (int j = 0; j < 8; ++j)
      m8[j] = fmaxf(m8[j], __shfl_xor(m8[j], d, 32));

  // ---- pass 2: exp / rowsum / PV ----
  v8f oc0 = {}, oc1 = {};
  float s8[8];
#pragma unroll
  for (int j = 0; j < 8; ++j) s8[j] = 0.f;

  for (int s0 = 0; s0 < SEQ; s0 += 32) {
    const __bf16* k0r = k_panel + (size_t)(s0 + nlo) * HDIM + half * 16;
    const __bf16* k1r = k0r + (size_t)16 * HDIM;
    v16bf bk0 = ldfrag(k0r, k0r + 8);
    v16bf bk1 = ldfrag(k1r, k1r + 8);
    v8f c0 = {}, c1 = {};
    c0 = __builtin_amdgcn_wmma_f32_16x16x32_bf16(false, aq, false, bk0,
                                                 (short)0, c0, false, false);
    c1 = __builtin_amdgcn_wmma_f32_16x16x32_bf16(false, aq, false, bk1,
                                                 (short)0, c1, false, false);
    const float* mp = mask + (size_t)(t0 + half * 8) * SEQ + s0 + nlo;
#pragma unroll
    for (int j = 0; j < 8; ++j) {
      float p0 = __expf(c0[j] + mp[(size_t)j * SEQ]      - m8[j]);
      float p1 = __expf(c1[j] + mp[(size_t)j * SEQ + 16] - m8[j]);
      s8[j] += p0 + p1;
      int r = j + half * 8;                        // C row for this element
      p_tile[r * 32 + nlo]      = (__bf16)p0;
      p_tile[r * 32 + 16 + nlo] = (__bf16)p1;
    }
    asm volatile("s_wait_dscnt 0" ::: "memory");   // same-wave LDS is in-order

    v16bf ap = ldfrag(p_tile + nlo * 32 + half * 16,
                      p_tile + nlo * 32 + half * 16 + 8);
    const __bf16* v0p = vbuf + ((size_t)b * EMB + head * HDIM + nlo) * SEQ
                       + s0 + half * 16;
    const __bf16* v1p = v0p + (size_t)16 * SEQ;
    v16bf bv0 = ldfrag(v0p, v0p + 8);
    v16bf bv1 = ldfrag(v1p, v1p + 8);
    oc0 = __builtin_amdgcn_wmma_f32_16x16x32_bf16(false, ap, false, bv0,
                                                  (short)0, oc0, false, false);
    oc1 = __builtin_amdgcn_wmma_f32_16x16x32_bf16(false, ap, false, bv1,
                                                  (short)0, oc1, false, false);
  }

#pragma unroll
  for (int d = 1; d < 16; d <<= 1)
#pragma unroll
    for (int j = 0; j < 8; ++j)
      s8[j] += __shfl_xor(s8[j], d, 32);

  v8bf r0, r1;
#pragma unroll
  for (int j = 0; j < 8; ++j) {
    float inv = 1.f / s8[j];
    r0[j] = (__bf16)(oc0[j] * inv);
    r1[j] = (__bf16)(oc1[j] * inv);
  }
  const size_t tbase = (size_t)t0 + 8 * half;     // rows contiguous per lane
  *(v8bf*)(abuf + ((size_t)b * EMB + head * HDIM + nlo)      * SEQ + tbase) = r0;
  *(v8bf*)(abuf + ((size_t)b * EMB + head * HDIM + 16 + nlo) * SEQ + tbase) = r1;
}

// ---------------------------------------------------------------------------
extern "C" void kernel_launch(void* const* d_in, const int* in_sizes, int n_in,
                              void* d_out, int out_size, void* d_ws, size_t ws_size,
                              hipStream_t stream) {
  (void)in_sizes; (void)n_in; (void)out_size; (void)ws_size;
  const float* query = (const float*)d_in[0];
  const float* key   = (const float*)d_in[1];
  const float* value = (const float*)d_in[2];
  const float* mask  = (const float*)d_in[3];
  const float* Wq = (const float*)d_in[4];
  const float* Wk = (const float*)d_in[5];
  const float* Wv = (const float*)d_in[6];
  const float* Wo = (const float*)d_in[7];
  const float* bq = (const float*)d_in[8];
  const float* bk = (const float*)d_in[9];
  const float* bv = (const float*)d_in[10];
  const float* bo = (const float*)d_in[11];

  const size_t NBH = (size_t)8 * NHEADS;            // 64
  __bf16* qbuf = (__bf16*)d_ws;                     // [64][1024][32]
  __bf16* kbuf = qbuf + NBH * SEQ * HDIM;           // [64][1024][32]
  __bf16* vbuf = kbuf + NBH * SEQ * HDIM;           // [8][256][1024]
  __bf16* abuf = vbuf + (size_t)8 * EMB * SEQ;      // [8][256][1024]

  dim3 gp(SEQ / 512, EMB / 16, 8);                  // (2,16,8)
  dim3 bp(256);
  const float scale = 0.17677669529663687f;         // 32^-0.5

  proj_gemm<<<gp, bp, 0, stream>>>(Wq, bq, query, 0, scale, 0, (void*)qbuf);
  proj_gemm<<<gp, bp, 0, stream>>>(Wk, bk, key,   0, 1.0f,  1, (void*)kbuf);
  proj_gemm<<<gp, bp, 0, stream>>>(Wv, bv, value, 0, 1.0f,  2, (void*)vbuf);

  const size_t attn_lds = (size_t)(32768 + 8 * 512) * sizeof(__bf16); // 73728B
  attn_kernel<<<dim3(SEQ / 128, (unsigned)NBH), 256, attn_lds, stream>>>(
      qbuf, kbuf, vbuf, mask, abuf);

  proj_gemm<<<gp, bp, 0, stream>>>(Wo, bo, abuf, 1, 1.0f, 3, d_out);
}
